// MultiHead_4088808865856
// MI455X (gfx1250) — matrix operand order repaired and checked by reading, not compiled, not run on hardware
//
#include <hip/hip_runtime.h>
#include <hip/hip_bf16.h>

// Problem constants (from reference)
#define BB 2
#define SS 2048
#define EE 1024
#define HH 16
#define DD 64
#define MM (BB*SS)   // 4096 rows in the flattened GEMMs

typedef __attribute__((ext_vector_type(16))) _Float16 v16h;
typedef __attribute__((ext_vector_type(8)))  _Float16 v8h;
typedef __attribute__((ext_vector_type(2)))  _Float16 v2h;
typedef __attribute__((ext_vector_type(8)))  float    v8f;
typedef __attribute__((ext_vector_type(4)))  float    v4f;

union AFrag { v16h v; v8h h8[2]; };

__device__ __forceinline__ v8f wmma_f16(v16h a, v16h b, v8f c) {
  // D = A(16x32 f16) * B(32x16 f16) + C(16x16 f32)
  return __builtin_amdgcn_wmma_f32_16x16x32_f16(false, a, false, b,
                                                (short)0, c, false, false);
}

// ---------------------------------------------------------------------------
// f32 -> f16 elementwise conversion (weights + activations)
// ---------------------------------------------------------------------------
__global__ __launch_bounds__(256) void cvt_f32_f16_kernel(
    const float* __restrict__ src, _Float16* __restrict__ dst, int n) {
  int i = (blockIdx.x * 256 + threadIdx.x) * 4;
  if (i + 3 < n) {
    v4f x = *(const v4f*)(src + i);
    v2h a, b;
    a.x = (_Float16)x.x; a.y = (_Float16)x.y;
    b.x = (_Float16)x.z; b.y = (_Float16)x.w;
    *(v2h*)(dst + i)     = a;
    *(v2h*)(dst + i + 2) = b;
  }
}

// ---------------------------------------------------------------------------
// QKV projection: Out = X16 @ W16^T + b, head-split stores. LDS-free:
// A and B fragments both stream from L2-resident f16 global buffers.
//   mode 0: Q, scaled by 1/sqrt(64), store [B,H,S,D]
//   mode 1: K, store [B,H,S,D]
//   mode 2: V, store transposed [B,H,D,S] (paired b32 stores along s)
// WG tile: 128(M) x 128(N); 8 waves: 4 along M x 2 along N,
// each wave 32x64 = 2x4 WMMA tiles, 8 WMMAs per 32-wide K-step.
// ---------------------------------------------------------------------------
__global__ __launch_bounds__(256) void qkv_proj_kernel(
    const _Float16* __restrict__ X16,   // [MM, EE] f16
    const _Float16* __restrict__ W16,   // [EE, EE] f16 row-major (n,k)
    const float* __restrict__ bias,     // [EE]
    _Float16* __restrict__ Out,
    int mode) {
  const int tid  = threadIdx.x;
  const int wid  = tid >> 5;
  const int lane = tid & 31;
  const int l16  = lane & 15;
  const int g    = lane >> 4;

  const int m0 = blockIdx.y * 128;
  const int n0 = blockIdx.x * 128;
  const int wm = (wid & 3) * 32;   // 4 waves along M
  const int wn = (wid >> 2) * 64;  // 2 waves along N

  v8f acc[2][4] = {};

  for (int k0 = 0; k0 < EE; k0 += 32) {
    // B fragments: 16 contiguous halves of W row n at column k0 + g*16
    // (contiguous k order, matching A).
    v16h bfrag[4];
#pragma unroll
    for (int ni = 0; ni < 4; ++ni) {
      const _Float16* bp =
          W16 + (size_t)(n0 + wn + ni * 16 + l16) * EE + k0 + g * 16;
      bfrag[ni] = *(const v16h*)bp;
    }
#pragma unroll
    for (int mi = 0; mi < 2; ++mi) {
      // A fragment: lane row = l16, two contiguous 8-half chunks at
      // k = g*16 and k = g*16 + 8 (contiguous order, matching B).
      const _Float16* ap =
          X16 + (size_t)(m0 + wm + mi * 16 + l16) * EE + k0;
      AFrag af;
      af.h8[0] = *(const v8h*)(ap + g * 16);
      af.h8[1] = *(const v8h*)(ap + g * 16 + 8);
#pragma unroll
      for (int ni = 0; ni < 4; ++ni)
        acc[mi][ni] = wmma_f16(af.v, bfrag[ni], acc[mi][ni]);
    }
  }

  // Store. C layout: lane holds rows r+8g (r=0..7), column n0+wn+ni*16+l16.
#pragma unroll
  for (int mi = 0; mi < 2; ++mi) {
#pragma unroll
    for (int ni = 0; ni < 4; ++ni) {
      const int ncol = n0 + wn + ni * 16 + l16;
      const float bv = bias[ncol];
      const int h = ncol >> 6;    // / DD
      const int d = ncol & 63;
      const int mbase = m0 + wm + mi * 16 + 8 * g;  // rows mbase..mbase+7
      const int b = mbase >> 11;                    // tile never crosses b
      const int sbase = mbase & 2047;
      if (mode == 2) {
        // V transposed: consecutive r -> consecutive s; pack pairs (b32).
        _Float16* op = Out + (((size_t)b * HH + h) * DD + d) * SS + sbase;
#pragma unroll
        for (int r = 0; r < 8; r += 2) {
          v2h pr;
          pr.x = (_Float16)(acc[mi][ni][r] + bv);
          pr.y = (_Float16)(acc[mi][ni][r + 1] + bv);
          *(v2h*)(op + r) = pr;
        }
      } else {
        const float scale = (mode == 0) ? 0.125f : 1.0f;
        _Float16* op = Out + (((size_t)b * HH + h) * SS + sbase) * DD + d;
#pragma unroll
        for (int r = 0; r < 8; ++r)
          op[(size_t)r * DD] = (_Float16)((acc[mi][ni][r] + bv) * scale);
      }
    }
  }
}

// ---------------------------------------------------------------------------
// Flash attention: one WG per (b, h, 128 q-rows); each wave owns 16 q-rows
// and streams K/V in chunks of 64 columns with online softmax.
// ---------------------------------------------------------------------------
#define KC 64               // kv columns per chunk
#define PS_STRIDE (KC + 8)  // 72 halves = 144B rows, keeps 16B alignment

__global__ __launch_bounds__(256) void attn_kernel(
    const _Float16* __restrict__ Qh,  // [B,H,S,D] (pre-scaled by 1/8)
    const _Float16* __restrict__ Kh,  // [B,H,S,D]
    const _Float16* __restrict__ Vt,  // [B,H,D,S]
    const unsigned char* __restrict__ mask,  // [B,S] bool
    _Float16* __restrict__ AO) {             // [B,S,E] f16
  __shared__ alignas(32) _Float16 Ps[8][16][PS_STRIDE];

  const int tid  = threadIdx.x;
  const int wid  = tid >> 5;
  const int lane = tid & 31;
  const int l16  = lane & 15;
  const int g    = lane >> 4;

  const int b  = blockIdx.z;
  const int h  = blockIdx.y;
  const int q0 = blockIdx.x * 128 + wid * 16;

  const _Float16* Qb = Qh + ((size_t)b * HH + h) * SS * DD;
  const _Float16* Kb = Kh + ((size_t)b * HH + h) * SS * DD;
  const _Float16* Vb = Vt + ((size_t)b * HH + h) * DD * SS;

  // Q A-fragments for both D k-steps, held for the whole kernel.
  AFrag qf[2];
#pragma unroll
  for (int ks = 0; ks < 2; ++ks) {
    const _Float16* qp = Qb + (size_t)(q0 + l16) * DD + ks * 32;
    qf[ks].h8[0] = *(const v8h*)(qp + g * 16);
    qf[ks].h8[1] = *(const v8h*)(qp + g * 16 + 8);
  }

  v8f o[4] = {};           // O accumulator, 4 d-tiles of 16
  float mrow[8], lrow[8];  // per-row softmax state (rows r+8g)
#pragma unroll
  for (int r = 0; r < 8; ++r) { mrow[r] = -3.0e38f; lrow[r] = 0.0f; }

  for (int s0 = 0; s0 < SS; s0 += KC) {
    // ---- scores: 4 column tiles x 2 k-steps over D ----
    v8f sc[4];
#pragma unroll
    for (int t = 0; t < 4; ++t) {
      v8f c = {};
#pragma unroll
      for (int ks = 0; ks < 2; ++ks) {
        const _Float16* kp =
            Kb + (size_t)(s0 + t * 16 + l16) * DD + ks * 32 + g * 16;
        v16h bf = *(const v16h*)kp;
        c = wmma_f16(qf[ks].v, bf, c);
      }
      sc[t] = c;
    }

    // Prefetch next chunk's K/V rows (uniform condition, no divergence);
    // overlaps with the softmax VALU below.
    if (s0 + KC < SS) {
      const _Float16* kn = Kb + (size_t)(s0 + KC + lane) * DD;
      __builtin_prefetch(kn, 0, 0);
      __builtin_prefetch(kn + (size_t)32 * DD, 0, 0);
      const _Float16* vn = Vb + (size_t)lane * SS + s0 + KC;
      __builtin_prefetch(vn, 0, 0);
      __builtin_prefetch(vn + (size_t)32 * SS, 0, 0);
    }

    // ---- boolean mask -> -inf (branchless select, EXEC stays full) ----
#pragma unroll
    for (int t = 0; t < 4; ++t) {
      const float lim = mask[(size_t)b * SS + s0 + t * 16 + l16] ? -3.0e38f
                                                                 : 3.0e38f;
#pragma unroll
      for (int r = 0; r < 8; ++r) sc[t][r] = fminf(sc[t][r], lim);
    }
    // ---- online softmax ----
    float mnew[8];
#pragma unroll
    for (int r = 0; r < 8; ++r) {
      float mx = fmaxf(fmaxf(sc[0][r], sc[1][r]), fmaxf(sc[2][r], sc[3][r]));
#pragma unroll
      for (int off = 1; off < 16; off <<= 1)
        mx = fmaxf(mx, __shfl_xor(mx, off, 32));
      mnew[r] = fmaxf(mrow[r], mx);
    }
    float rsum[8];
#pragma unroll
    for (int r = 0; r < 8; ++r) rsum[r] = 0.0f;
#pragma unroll
    for (int t = 0; t < 4; ++t) {
#pragma unroll
      for (int r = 0; r < 8; ++r) {
        const float p = __expf(sc[t][r] - mnew[r]);
        rsum[r] += p;
        Ps[wid][r + 8 * g][t * 16 + l16] = (_Float16)p;  // C-layout -> LDS
      }
    }
#pragma unroll
    for (int r = 0; r < 8; ++r) {
      float sm = rsum[r];
#pragma unroll
      for (int off = 1; off < 16; off <<= 1) sm += __shfl_xor(sm, off, 32);
      const float alpha = __expf(mrow[r] - mnew[r]);
      lrow[r] = lrow[r] * alpha + sm;
      mrow[r] = mnew[r];
#pragma unroll
      for (int t = 0; t < 4; ++t) o[t][r] *= alpha;
    }
    // LDS write->read is per-wave (in-order); make the dependency explicit.
    asm volatile("s_wait_dscnt 0" ::: "memory");

    // ---- O += P @ V : read P back in A-fragment layout ----
    AFrag pf[2];
#pragma unroll
    for (int ks = 0; ks < 2; ++ks) {
      const _Float16* pp = &Ps[wid][l16][0];
      pf[ks].h8[0] = *(const v8h*)(pp + ks * 32 + g * 16);
      pf[ks].h8[1] = *(const v8h*)(pp + ks * 32 + g * 16 + 8);
    }
#pragma unroll
    for (int t = 0; t < 4; ++t) {  // d-tiles
#pragma unroll
      for (int ks = 0; ks < 2; ++ks) {
        const _Float16* vp =
            Vb + (size_t)(t * 16 + l16) * SS + s0 + ks * 32 + g * 16;
        v16h bf = *(const v16h*)vp;
        o[t] = wmma_f16(pf[ks].v, bf, o[t]);
      }
    }
  }

  // ---- finalize: O /= l, store f16 into [B,S,E] for the output GEMM ----
#pragma unroll
  for (int r = 0; r < 8; ++r) lrow[r] = 1.0f / lrow[r];
#pragma unroll
  for (int t = 0; t < 4; ++t) {
#pragma unroll
    for (int r = 0; r < 8; ++r) {
      const int srow = q0 + r + 8 * g;
      const int col  = h * DD + t * 16 + l16;
      AO[((size_t)b * SS + srow) * EE + col] = (_Float16)(o[t][r] * lrow[r]);
    }
  }
}

// ---------------------------------------------------------------------------
// Output projection: out = attn @ Wo^T + bo + residual  (f32 result)
// WG tile 128x128, wave tile 32x64, LDS-free.
// ---------------------------------------------------------------------------
__global__ __launch_bounds__(256) void out_proj_kernel(
    const _Float16* __restrict__ A16,   // [MM, EE] f16
    const _Float16* __restrict__ Wo16,  // [EE, EE] f16
    const float* __restrict__ bo,       // [EE]
    const float* __restrict__ residual, // [MM, EE] f32 (q input)
    float* __restrict__ Out) {          // [MM, EE] f32
  const int tid  = threadIdx.x;
  const int wid  = tid >> 5;
  const int lane = tid & 31;
  const int l16  = lane & 15;
  const int g    = lane >> 4;

  const int m0 = blockIdx.y * 128;
  const int n0 = blockIdx.x * 128;
  const int wm = (wid & 3) * 32;
  const int wn = (wid >> 2) * 64;

  v8f acc[2][4] = {};

  for (int k0 = 0; k0 < EE; k0 += 32) {
    v16h bfrag[4];
#pragma unroll
    for (int ni = 0; ni < 4; ++ni) {
      const _Float16* bp =
          Wo16 + (size_t)(n0 + wn + ni * 16 + l16) * EE + k0 + g * 16;
      bfrag[ni] = *(const v16h*)bp;
    }
#pragma unroll
    for (int mi = 0; mi < 2; ++mi) {
      const _Float16* ap = A16 + (size_t)(m0 + wm + mi * 16 + l16) * EE + k0;
      AFrag af;
      af.h8[0] = *(const v8h*)(ap + g * 16);
      af.h8[1] = *(const v8h*)(ap + g * 16 + 8);
#pragma unroll
      for (int ni = 0; ni < 4; ++ni)
        acc[mi][ni] = wmma_f16(af.v, bfrag[ni], acc[mi][ni]);
    }
  }

#pragma unroll
  for (int mi = 0; mi < 2; ++mi) {
#pragma unroll
    for (int ni = 0; ni < 4; ++ni) {
      const int ncol = n0 + wn + ni * 16 + l16;
      const float bb = bo[ncol];
#pragma unroll
      for (int r = 0; r < 8; ++r) {
        const int mrow = m0 + wm + mi * 16 + r + 8 * g;
        const size_t idx = (size_t)mrow * EE + ncol;
        Out[idx] = acc[mi][ni][r] + bb + residual[idx];
      }
    }
  }
}

// ---------------------------------------------------------------------------
// Host launch
// ---------------------------------------------------------------------------
extern "C" void kernel_launch(void* const* d_in, const int* in_sizes, int n_in,
                              void* d_out, int out_size, void* d_ws,
                              size_t ws_size, hipStream_t stream) {
  (void)in_sizes; (void)n_in; (void)out_size; (void)ws_size;
  const float* q    = (const float*)d_in[0];
  const float* k    = (const float*)d_in[1];
  const float* v    = (const float*)d_in[2];
  const unsigned char* mask = (const unsigned char*)d_in[3];
  const float* Wq = (const float*)d_in[4];
  const float* bq = (const float*)d_in[5];
  const float* Wk = (const float*)d_in[6];
  const float* bk = (const float*)d_in[7];
  const float* Wv = (const float*)d_in[8];
  const float* bv = (const float*)d_in[9];
  const float* Wo = (const float*)d_in[10];
  const float* bo = (const float*)d_in[11];
  float* out = (float*)d_out;

  // Workspace layout (f16):
  //   [0,8MB)    : Wq16..Wo16 (2MB each)
  //   [8,32MB)   : Qh, Kh, Vt (8MB each)
  //   [32,56MB)  : X16q, X16k, X16v (8MB each)
  //   AO aliases X16q (dead once the qkv GEMMs complete).
  char* ws = (char*)d_ws;
  const size_t WSZ = (size_t)EE * EE * sizeof(_Float16);            // 2 MB
  const size_t TSZ = (size_t)BB * HH * SS * DD * sizeof(_Float16);  // 8 MB
  _Float16* Wq16 = (_Float16*)(ws);
  _Float16* Wk16 = (_Float16*)(ws + WSZ);
  _Float16* Wv16 = (_Float16*)(ws + 2 * WSZ);
  _Float16* Wo16 = (_Float16*)(ws + 3 * WSZ);
  _Float16* Qh   = (_Float16*)(ws + 4 * WSZ);
  _Float16* Kh   = (_Float16*)(ws + 4 * WSZ + TSZ);
  _Float16* Vt   = (_Float16*)(ws + 4 * WSZ + 2 * TSZ);
  _Float16* X16q = (_Float16*)(ws + 4 * WSZ + 3 * TSZ);
  _Float16* X16k = (_Float16*)(ws + 4 * WSZ + 4 * TSZ);
  _Float16* X16v = (_Float16*)(ws + 4 * WSZ + 5 * TSZ);
  _Float16* AO   = X16q;  // alias: X16* dead before attention runs

  // 1) f32 -> f16 conversions (weights + activations)
  const int nW = EE * EE;
  const int wblk = nW / 4 / 256;          // 1024
  cvt_f32_f16_kernel<<<wblk, 256, 0, stream>>>(Wq, Wq16, nW);
  cvt_f32_f16_kernel<<<wblk, 256, 0, stream>>>(Wk, Wk16, nW);
  cvt_f32_f16_kernel<<<wblk, 256, 0, stream>>>(Wv, Wv16, nW);
  cvt_f32_f16_kernel<<<wblk, 256, 0, stream>>>(Wo, Wo16, nW);
  const int nX = MM * EE;
  const int xblk = nX / 4 / 256;          // 4096
  cvt_f32_f16_kernel<<<xblk, 256, 0, stream>>>(q, X16q, nX);
  cvt_f32_f16_kernel<<<xblk, 256, 0, stream>>>(k, X16k, nX);
  cvt_f32_f16_kernel<<<xblk, 256, 0, stream>>>(v, X16v, nX);

  // 2) QKV projections (LDS-free WMMA GEMMs)
  dim3 gp(EE / 128, MM / 128);  // (8, 32)
  qkv_proj_kernel<<<gp, 256, 0, stream>>>(X16q, Wq16, bq, Qh, 0);
  qkv_proj_kernel<<<gp, 256, 0, stream>>>(X16k, Wk16, bk, Kh, 1);
  qkv_proj_kernel<<<gp, 256, 0, stream>>>(X16v, Wv16, bv, Vt, 2);

  // 3) flash attention
  attn_kernel<<<dim3(SS / 128, HH, BB), 256, 0, stream>>>(Qh, Kh, Vt, mask, AO);

  // 4) output projection + residual
  out_proj_kernel<<<dim3(EE / 128, MM / 128), 256, 0, stream>>>(AO, Wo16, bo, q,
                                                                out);
}
